// CrossSparseAggrNet_v2_38208029065656
// MI455X (gfx1250) — matrix operand order repaired and checked by reading, not compiled, not run on hardware
//
#include <hip/hip_runtime.h>
#include <hip/hip_bf16.h>

// ---------------------------------------------------------------------------
// Problem constants (from reference)
// ---------------------------------------------------------------------------
#define B_V 32
#define L_V 196
#define CH  512
#define B_T 32
#define NUM_KEEP 118
#define NUM_NON  78      // 196 - 118
#define KEEPED   47
#define HID      102
#define MP  128          // padded keep rows (118 -> 128)
#define HP  128          // padded hidden  (102 -> 128)
#define KP  48           // toks (47 aggr + 1 extra)
#define WP  80           // padded caption words (<=70 -> 80)

// ---------------------------------------------------------------------------
// WMMA types & fragment helpers (CDNA5 bf16 16x16x32, wave32)
// ---------------------------------------------------------------------------
typedef __attribute__((ext_vector_type(16))) __bf16 v16bf;
typedef __attribute__((ext_vector_type(8)))  float  v8f;

union FragU { uint4 q[2]; v16bf v; };

__device__ __forceinline__ unsigned short f2bf(float f) {
  unsigned int u = __float_as_uint(f);
  unsigned int r = u + 0x7fffu + ((u >> 16) & 1u);   // round-to-nearest-even
  return (unsigned short)(r >> 16);
}

// A matrix (M x K, 16-bit, row-major, ld = elements per row).
// Lane l<16: row m0+l, elems[0..7]=K k0..k0+7, elems[8..15]=K k0+16..k0+23.
// Lane l>=16: row m0+l-16, runs shifted by +8 in K.
__device__ __forceinline__ v16bf load_a16(const unsigned short* base, int m0,
                                          int k0, int ld, int lane) {
  int row = m0 + (lane & 15);
  int off = (lane >> 4) << 3;                 // 0 or 8
  const unsigned short* p = base + row * ld + k0 + off;
  FragU u;
  u.q[0] = *(const uint4*)(p);                // K .. K+7
  u.q[1] = *(const uint4*)(p + 16);           // K+16 .. K+23
  return u.v;
}

// B matrix (K x N) stored column-major as [n][k] (ld = K stride).
// Lane l<16: col n0+l, elems[0..7]=K k0..k0+7, elems[8..15]=K k0+16..k0+23. Lane l>=16: shifted by +8 in K.
__device__ __forceinline__ v16bf load_b16(const unsigned short* base, int n0,
                                          int k0, int ld, int lane) {
  int col = n0 + (lane & 15);
  int off = (lane >> 4) << 3;                 // 0 or 8
  const unsigned short* p = base + col * ld + k0 + off;
  FragU u;
  u.q[0] = *(const uint4*)(p);
  u.q[1] = *(const uint4*)(p + 16);
  return u.v;
}

__device__ __forceinline__ v8f wmma_bf16(v16bf a, v16bf b, v8f c) {
  return __builtin_amdgcn_wmma_f32_16x16x32_bf16(false, a, false, b,
                                                 (short)0, c, false, false);
}

__device__ __forceinline__ float wave_sum(float v) {
  for (int off = 16; off > 0; off >>= 1) v += __shfl_xor(v, off, 32);
  return v;
}

// ---------------------------------------------------------------------------
// K1: per-image stats: token inverse L2 norms + self attention (glo.img_norm)
// ---------------------------------------------------------------------------
__global__ void img_stats_kernel(const float* __restrict__ img,
                                 float* __restrict__ invn,
                                 float* __restrict__ sattn) {
  __shared__ float gs[CH];
  __shared__ float red[256];
  const int b = blockIdx.x;
  const int tid = threadIdx.x;
  const float* base = img + (size_t)b * L_V * CH;

  // per-channel mean over 196 tokens
  for (int c = tid; c < CH; c += 256) {
    float s = 0.f;
    for (int l = 0; l < L_V; ++l) s += base[(size_t)l * CH + c];
    gs[c] = s * (1.f / (float)L_V);
  }
  __syncthreads();
  float sq = 0.f;
  for (int c = tid; c < CH; c += 256) sq += gs[c] * gs[c];
  red[tid] = sq;
  __syncthreads();
  for (int s = 128; s > 0; s >>= 1) {
    if (tid < s) red[tid] += red[tid + s];
    __syncthreads();
  }
  if (tid == 0) red[0] = 1.f / fmaxf(sqrtf(red[0]), 1e-12f);
  __syncthreads();
  float inv = red[0];
  for (int c = tid; c < CH; c += 256) gs[c] *= inv;
  __syncthreads();

  // per-token inverse norm and self-attn score
  for (int l = tid; l < L_V; l += 256) {
    const float* xr = base + (size_t)l * CH;
    float ss = 0.f, dg = 0.f;
    for (int c = 0; c < CH; ++c) {
      float x = xr[c];
      ss += x * x;
      dg += gs[c] * x;
    }
    float iv = 1.f / fmaxf(sqrtf(ss), 1e-12f);
    invn[b * L_V + l] = iv;
    sattn[b * L_V + l] = dg * iv;
  }
}

// ---------------------------------------------------------------------------
// K2: caption stats: cap_glo (masked mean, l2-normed) + per-word inv norms
// ---------------------------------------------------------------------------
__global__ void cap_stats_kernel(const float* __restrict__ cap,
                                 const int* __restrict__ lens, int Lcap,
                                 float* __restrict__ capglo,
                                 float* __restrict__ invwn) {
  __shared__ float cs[CH];
  __shared__ float red[256];
  const int b = blockIdx.x;
  const int tid = threadIdx.x;
  const int len = lens[b];
  const float nf = (float)len;
  const float* base = cap + (size_t)b * Lcap * CH;

  for (int c = tid; c < CH; c += 256) {
    float s = 0.f;
    for (int w = 0; w < Lcap; ++w)
      if (w < len) s += base[(size_t)w * CH + c];
    cs[c] = s / nf;
  }
  __syncthreads();
  float sq = 0.f;
  for (int c = tid; c < CH; c += 256) sq += cs[c] * cs[c];
  red[tid] = sq;
  __syncthreads();
  for (int s = 128; s > 0; s >>= 1) {
    if (tid < s) red[tid] += red[tid + s];
    __syncthreads();
  }
  if (tid == 0) red[0] = 1.f / fmaxf(sqrtf(red[0]), 1e-12f);
  __syncthreads();
  float inv = red[0];
  for (int c = tid; c < CH; c += 256) capglo[b * CH + c] = cs[c] * inv;

  for (int w = tid; w < Lcap; w += 256) {
    const float* xr = base + (size_t)w * CH;
    float ss = 0.f;
    for (int c = 0; c < CH; ++c) ss += xr[c] * xr[c];
    invwn[b * WP + w] = 1.f / fmaxf(sqrtf(ss), 1e-12f);
  }
}

// ---------------------------------------------------------------------------
// K3: weight prep: W1 (512x102) -> bf16 col-major [128][512] (zero padded)
//                  W2 (102x47)  -> bf16 col-major [48][128]
// ---------------------------------------------------------------------------
__global__ void prep_w_kernel(const float* __restrict__ w1,
                              const float* __restrict__ w2,
                              unsigned short* __restrict__ w1t,
                              unsigned short* __restrict__ w2t) {
  int idx = blockIdx.x * blockDim.x + threadIdx.x;
  if (idx < HP * CH) {
    int n = idx / CH, k = idx % CH;
    float v = (n < HID) ? w1[(size_t)k * HID + n] : 0.f;
    w1t[idx] = f2bf(v);
  } else if (idx < HP * CH + KP * HP) {
    int j = idx - HP * CH;
    int n = j / HP, k = j % HP;
    float v = (n < KEEPED && k < HID) ? w2[(size_t)k * KEEPED + n] : 0.f;
    w2t[j] = f2bf(v);
  }
}

__global__ void zero_out_kernel(float* __restrict__ out) {
  int idx = blockIdx.x * blockDim.x + threadIdx.x;
  if (idx < B_V * B_T) out[idx] = 0.f;
}

// ---------------------------------------------------------------------------
// K4: per (caption, image) pair: score -> bitonic sort -> keep idx + pooled
//     "extra" token from the 78 non-kept tokens (softmax weighted).
// ---------------------------------------------------------------------------
__global__ void score_sort_kernel(const float* __restrict__ img,
                                  const float* __restrict__ invn,
                                  const float* __restrict__ sattn,
                                  const float* __restrict__ capglo,
                                  int* __restrict__ keep_ws,
                                  float* __restrict__ extra_ws) {
  __shared__ float cg[CH];
  __shared__ float skey[256];
  __shared__ int   sidx[256];
  __shared__ float nw[NUM_NON];
  __shared__ float inv_sum;

  const int bv = blockIdx.x, bt = blockIdx.y;
  const int tid = threadIdx.x;
  const size_t pair = (size_t)bt * B_V + bv;

  for (int c = tid; c < CH; c += 256) cg[c] = capglo[bt * CH + c];
  __syncthreads();

  if (tid < L_V) {
    const float* xr = img + ((size_t)bv * L_V + tid) * CH;
    float d = 0.f;
    for (int c = 0; c < CH; ++c) d += cg[c] * xr[c];
    skey[tid] = sattn[bv * L_V + tid] + d * invn[bv * L_V + tid];
    sidx[tid] = tid;
  } else {
    skey[tid] = -1e30f;
    sidx[tid] = 0;
  }

  // bitonic sort, descending
  for (int k = 2; k <= 256; k <<= 1) {
    for (int j = k >> 1; j > 0; j >>= 1) {
      __syncthreads();
      int i = tid, ixj = i ^ j;
      if (ixj > i) {
        bool desc = ((i & k) == 0);
        float a = skey[i], b = skey[ixj];
        bool sw = desc ? (a < b) : (a > b);
        if (sw) {
          skey[i] = b; skey[ixj] = a;
          int t = sidx[i]; sidx[i] = sidx[ixj]; sidx[ixj] = t;
        }
      }
    }
  }
  __syncthreads();

  if (tid < MP) keep_ws[pair * MP + tid] = sidx[tid];

  // softmax over the 78 non-kept sorted scores
  if (tid < NUM_NON) nw[tid] = expf(skey[NUM_KEEP + tid] - skey[NUM_KEEP]);
  __syncthreads();
  if (tid == 0) {
    float s = 0.f;
    for (int i = 0; i < NUM_NON; ++i) s += nw[i];
    inv_sum = 1.f / s;
  }
  __syncthreads();

  for (int c = tid; c < CH; c += 256) {
    float acc = 0.f;
    for (int i = 0; i < NUM_NON; ++i) {
      int t = sidx[NUM_KEEP + i];
      acc += nw[i] * img[((size_t)bv * L_V + t) * CH + c];
    }
    extra_ws[pair * CH + c] = acc * inv_sum;
  }
}

// ---------------------------------------------------------------------------
// K5: fused per-pair: LN -> WMMA GEMM1 -> GELU -> WMMA GEMM2 -> softmax ->
//     aggregation -> l2norm -> WMMA xattn -> masked max/sum -> atomicAdd out
// Dynamic LDS map (bytes), phase-overlapped:
//   XBF    @ 0      : 128*512 bf16 (phase 1-2)
//   LOGITS @ 0      : 128*48  f32  (phase 3-5)
//   XATT   @ 0      : 80*48   f32  (phase 7-8)
//   TOKS   @ 24576  : 48*512  f32  (phase 5-6)
//   HBF    @ 131072 : 128*128 bf16 (phase 2-3)
//   CAPBF  @ 131072 : 80*512  bf16 (phase 6-7)
//   TOKSBF @ 212992 : 48*512  bf16 (phase 6-7)
//   RED    @ 262144 : 256 f32, KIDX @ 263168 : 128 i32  -> total 263680
// ---------------------------------------------------------------------------
#define SMEM_P3 263680

__global__ void pair_fused_kernel(const float* __restrict__ img,
                                  const float* __restrict__ cap,
                                  const int* __restrict__ lens,
                                  const float* __restrict__ invwn,
                                  const int* __restrict__ keep_ws,
                                  const float* __restrict__ extra_ws,
                                  const unsigned short* __restrict__ w1t,
                                  const unsigned short* __restrict__ w2t,
                                  const float* __restrict__ b1,
                                  const float* __restrict__ b2,
                                  const float* __restrict__ scale,
                                  const float* __restrict__ lng,
                                  const float* __restrict__ lnb,
                                  int Lcap, float* __restrict__ out) {
  extern __shared__ char smem[];
  unsigned short* XBF    = (unsigned short*)(smem + 0);
  float*          LOGITS = (float*)(smem + 0);
  float*          XATT   = (float*)(smem + 0);
  float*          TOKS   = (float*)(smem + 24576);
  unsigned short* HBF    = (unsigned short*)(smem + 131072);
  unsigned short* CAPBF  = (unsigned short*)(smem + 131072);
  unsigned short* TOKSBF = (unsigned short*)(smem + 212992);
  float*          RED    = (float*)(smem + 262144);
  int*            KIDX   = (int*)(smem + 263168);

  const int bv = blockIdx.x, bt = blockIdx.y;
  const int tid = threadIdx.x;
  const int wave = tid >> 5, lane = tid & 31;
  const size_t pair = (size_t)bt * B_V + bv;
  const float sc = scale[0];

  if (tid < MP) KIDX[tid] = keep_ws[pair * MP + tid];
  __syncthreads();

  // ---- phase 1: gather selected tokens + LayerNorm -> XBF (bf16) ----
  for (int row = wave; row < MP; row += 8) {
    const float* xr = img + ((size_t)bv * L_V + KIDX[row]) * CH;
    float vals[16], s = 0.f, s2 = 0.f;
    for (int j = 0; j < 16; ++j) {
      float x = xr[lane + 32 * j];
      vals[j] = x; s += x; s2 += x * x;
    }
    s  = wave_sum(s);
    s2 = wave_sum(s2);
    float mu   = s * (1.f / (float)CH);
    float var  = s2 * (1.f / (float)CH) - mu * mu;
    float rstd = rsqrtf(var + 1e-5f);
    for (int j = 0; j < 16; ++j) {
      int c = lane + 32 * j;
      XBF[row * CH + c] = f2bf((vals[j] - mu) * rstd * lng[c] + lnb[c]);
    }
  }
  __syncthreads();

  // ---- phase 2: GEMM1  H = gelu(X @ W1 + b1), 128x128x512 (WMMA) ----
  {
    const int m0 = 16 * wave;
    for (int nt = 0; nt < HP / 16; ++nt) {
      v8f acc = {0.f, 0.f, 0.f, 0.f, 0.f, 0.f, 0.f, 0.f};
      for (int kt = 0; kt < CH / 32; ++kt) {
        v16bf a = load_a16(XBF, m0, 32 * kt, CH, lane);
        v16bf b = load_b16(w1t, 16 * nt, 32 * kt, CH, lane);
        acc = wmma_bf16(a, b, acc);
      }
      int col = 16 * nt + (lane & 15);
      float bias = (col < HID) ? b1[col] : 0.f;
      int roff = (lane >> 4) << 3;
      for (int r = 0; r < 8; ++r) {
        int row = m0 + r + roff;
        float x = acc[r] + bias;
        float g = 0.5f * x * (1.f + erff(x * 0.70710678118f));  // exact gelu
        HBF[row * HP + col] = f2bf(g);
      }
    }
  }
  __syncthreads();

  // ---- phase 3: GEMM2  logits = (H @ W2 + b2) * scale, 128x48x128 ----
  {
    const int m0 = 16 * wave;
    for (int nt = 0; nt < KP / 16; ++nt) {
      v8f acc = {0.f, 0.f, 0.f, 0.f, 0.f, 0.f, 0.f, 0.f};
      for (int kt = 0; kt < HP / 32; ++kt) {
        v16bf a = load_a16(HBF, m0, 32 * kt, HP, lane);
        v16bf b = load_b16(w2t, 16 * nt, 32 * kt, HP, lane);
        acc = wmma_bf16(a, b, acc);
      }
      int col = 16 * nt + (lane & 15);
      float bias = (col < KEEPED) ? b2[col] : 0.f;
      int roff = (lane >> 4) << 3;
      for (int r = 0; r < 8; ++r) {
        int row = m0 + r + roff;
        LOGITS[row * KP + col] = (acc[r] + bias) * sc;
      }
    }
  }
  __syncthreads();

  // ---- phase 4: softmax over the 118 kept tokens, per aggregated tok ----
  if (tid < KEEPED) {
    float m = -1e30f;
    for (int l = 0; l < NUM_KEEP; ++l) m = fmaxf(m, LOGITS[l * KP + tid]);
    float s = 0.f;
    for (int l = 0; l < NUM_KEEP; ++l) s += expf(LOGITS[l * KP + tid] - m);
    float is = 1.f / s;
    for (int l = 0; l < NUM_KEEP; ++l)
      LOGITS[l * KP + tid] = expf(LOGITS[l * KP + tid] - m) * is;
  }
  __syncthreads();

  // ---- phase 5: aggregation toks[k] = sum_l w[l,k]*sel_raw[l] + extra ----
  for (int k = wave; k < KEEPED; k += 8) {
    float acc[16];
    for (int j = 0; j < 16; ++j) acc[j] = 0.f;
    for (int l = 0; l < NUM_KEEP; ++l) {
      float wlk = LOGITS[l * KP + k];
      const float* xr = img + ((size_t)bv * L_V + KIDX[l]) * CH;
      if (l + 1 < NUM_KEEP)
        __builtin_prefetch(img + ((size_t)bv * L_V + KIDX[l + 1]) * CH + lane, 0, 1);
      for (int j = 0; j < 16; ++j) acc[j] += wlk * xr[lane + 32 * j];
    }
    for (int j = 0; j < 16; ++j) TOKS[k * CH + lane + 32 * j] = acc[j];
  }
  for (int c = tid; c < CH; c += 256)
    TOKS[KEEPED * CH + c] = extra_ws[pair * CH + c];
  __syncthreads();

  // ---- phase 6: l2norm rows -> TOKSBF; build normalized caption CAPBF ----
  for (int r = wave; r < KP; r += 8) {
    float vals[16], s2 = 0.f;
    for (int j = 0; j < 16; ++j) {
      vals[j] = TOKS[r * CH + lane + 32 * j];
      s2 += vals[j] * vals[j];
    }
    s2 = wave_sum(s2);
    float inv = 1.f / fmaxf(sqrtf(s2), 1e-12f);
    for (int j = 0; j < 16; ++j)
      TOKSBF[r * CH + lane + 32 * j] = f2bf(vals[j] * inv);
  }
  const int len = lens[bt];
  for (int idx = tid; idx < WP * CH; idx += 256) {
    int w = idx >> 9, c = idx & (CH - 1);
    float v = 0.f;
    if (w < len) v = cap[((size_t)bt * Lcap + w) * CH + c] * invwn[bt * WP + w];
    CAPBF[idx] = f2bf(v);
  }
  __syncthreads();

  // ---- phase 7: xattn = capn @ toks^T, 80x48x512 (WMMA) ----
  for (int t = wave; t < (WP / 16) * (KP / 16); t += 8) {
    int mt = t / (KP / 16), nt = t % (KP / 16);
    v8f acc = {0.f, 0.f, 0.f, 0.f, 0.f, 0.f, 0.f, 0.f};
    for (int kt = 0; kt < CH / 32; ++kt) {
      v16bf a = load_a16(CAPBF, 16 * mt, 32 * kt, CH, lane);
      v16bf b = load_b16(TOKSBF, 16 * nt, 32 * kt, CH, lane);
      acc = wmma_bf16(a, b, acc);
    }
    int col = 16 * nt + (lane & 15);
    int roff = (lane >> 4) << 3;
    for (int r = 0; r < 8; ++r)
      XATT[(16 * mt + r + roff) * KP + col] = acc[r];
  }
  __syncthreads();

  // ---- phase 8: sim = sum_words max_k xattn / len ; atomicAdd to out ----
  float v = 0.f;
  if (tid < WP && tid < len) {
    float mx = -1e30f;
    for (int k = 0; k < KP; ++k) mx = fmaxf(mx, XATT[tid * KP + k]);
    v = mx;
  }
  RED[tid] = v;
  __syncthreads();
  for (int s = 128; s > 0; s >>= 1) {
    if (tid < s) RED[tid] += RED[tid + s];
    __syncthreads();
  }
  if (tid == 0) atomicAdd(&out[bv * B_T + bt], RED[0] / (float)len);
}

// ---------------------------------------------------------------------------
// Host launcher
// ---------------------------------------------------------------------------
extern "C" void kernel_launch(void* const* d_in, const int* in_sizes, int n_in,
                              void* d_out, int out_size, void* d_ws,
                              size_t ws_size, hipStream_t stream) {
  (void)in_sizes; (void)n_in; (void)out_size; (void)ws_size;
  const float* img   = (const float*)d_in[0];
  const float* cap   = (const float*)d_in[1];
  const float* lcap  = (const float*)d_in[2];
  const float* lng   = (const float*)d_in[3];
  const float* lnb   = (const float*)d_in[4];
  const float* w1    = (const float*)d_in[5];
  const float* b1    = (const float*)d_in[6];
  const float* w2    = (const float*)d_in[7];
  const float* b2    = (const float*)d_in[8];
  const float* scale = (const float*)d_in[9];
  const int*   clen  = (const int*)d_in[10];
  const int*   llen  = (const int*)d_in[11];
  float* out = (float*)d_out;

  char* ws = (char*)d_ws;
  size_t o = 0;
  auto carve = [&](size_t bytes) -> char* {
    char* p = ws + o;
    o = (o + bytes + 255) & ~(size_t)255;
    return p;
  };
  float* invn  = (float*)carve((size_t)B_V * L_V * 4);
  float* sattn = (float*)carve((size_t)B_V * L_V * 4);
  unsigned short* w1t = (unsigned short*)carve((size_t)HP * CH * 2);
  unsigned short* w2t = (unsigned short*)carve((size_t)KP * HP * 2);
  float* cg0 = (float*)carve((size_t)B_T * CH * 4);
  float* cg1 = (float*)carve((size_t)B_T * CH * 4);
  float* iw0 = (float*)carve((size_t)B_T * WP * 4);
  float* iw1 = (float*)carve((size_t)B_T * WP * 4);
  int* keep0 = (int*)carve((size_t)B_T * B_V * MP * 4);
  int* keep1 = (int*)carve((size_t)B_T * B_V * MP * 4);
  float* ex0 = (float*)carve((size_t)B_T * B_V * CH * 4);
  float* ex1 = (float*)carve((size_t)B_T * B_V * CH * 4);

  img_stats_kernel<<<B_V, 256, 0, stream>>>(img, invn, sattn);
  cap_stats_kernel<<<B_T, 256, 0, stream>>>(cap, clen, 40, cg0, iw0);
  cap_stats_kernel<<<B_T, 256, 0, stream>>>(lcap, llen, 70, cg1, iw1);
  {
    int total = HP * CH + KP * HP;
    prep_w_kernel<<<(total + 255) / 256, 256, 0, stream>>>(w1, w2, w1t, w2t);
  }
  zero_out_kernel<<<(B_V * B_T + 255) / 256, 256, 0, stream>>>(out);

  dim3 g(B_V, B_T);
  score_sort_kernel<<<g, 256, 0, stream>>>(img, invn, sattn, cg0, keep0, ex0);
  score_sort_kernel<<<g, 256, 0, stream>>>(img, invn, sattn, cg1, keep1, ex1);

  pair_fused_kernel<<<g, 256, SMEM_P3, stream>>>(
      img, cap, clen, iw0, keep0, ex0, w1t, w2t, b1, b2, scale, lng, lnb, 40, out);
  pair_fused_kernel<<<g, 256, SMEM_P3, stream>>>(
      img, lcap, llen, iw1, keep1, ex1, w1t, w2t, b1, b2, scale, lng, lnb, 70, out);
}
